// LucyRNNCellTriton_22643067585253
// MI455X (gfx1250) — matrix operand order repaired and checked by reading, not compiled, not run on hardware
//
#include <hip/hip_runtime.h>
#include <hip/hip_bf16.h>
#include <stdint.h>

// ---------------------------------------------------------------------------
// LucyRNN cell on MI455X (gfx1250):
//   Phase 1: convert x -> bf16 (row-major [M,K]), W -> bf16 transposed [K,N]
//   Phase 2: gates = x @ W^T + b via v_wmma_f32_16x16x32_bf16 (f32 accum),
//            double-buffered LDS tiles filled with GLOBAL_LOAD_ASYNC_TO_LDS
//   Phase 3: fused activation + sequential scan over T, one thread per (b,d)
// Workspace layout (assumed large enough; ~495 MB):
//   [0, 32MiB)      Xb  bf16 [16384,1024]
//   [+14MiB)        Wt  bf16 [1024,7168]   (W transposed)
//   [+448MiB)       gates f32 [16384,7168]
// ---------------------------------------------------------------------------

typedef __attribute__((ext_vector_type(16))) __bf16 v16bf;
typedef __attribute__((ext_vector_type(8)))  float  v8f;
typedef __attribute__((ext_vector_type(4)))  int    v4i_t;

#define B_   8
#define T_   2048
#define K_   1024          // DIN
#define D_   1024
#define M_   (B_ * T_)     // 16384
#define N_   (7 * D_)      // 7168
#define EPS_ 1e-6f

#if __has_builtin(__builtin_amdgcn_global_load_async_to_lds_b128) && \
    __has_builtin(__builtin_amdgcn_s_wait_asynccnt)
#define USE_ASYNC_LDS 1
#else
#define USE_ASYNC_LDS 0
#endif

// Pointer-to-int4 in explicit address spaces (what the async builtin expects:
// param 1 is "int __vector(4) __device__ *" = v4i addrspace(1)*).
typedef __attribute__((address_space(1))) v4i_t* gas_v4i_p;
typedef __attribute__((address_space(3))) v4i_t* lds_v4i_p;

__device__ __forceinline__ unsigned short f32_to_bf16(float f) {
    unsigned int u = __float_as_uint(f);
    unsigned int rnd = 0x7FFFu + ((u >> 16) & 1u);   // round-to-nearest-even
    return (unsigned short)((u + rnd) >> 16);
}

// 16-byte global -> LDS copy; async (no VGPR staging, ASYNCcnt) when available.
__device__ __forceinline__ void copy16_g2l(const unsigned short* g,
                                           unsigned short* l) {
#if USE_ASYNC_LDS
    // inttoptr casts: global flat address == addrspace(1) address; low 32 bits
    // of a flat LDS address are the wave-relative LDS byte offset (ISA 10.2).
    __builtin_amdgcn_global_load_async_to_lds_b128(
        (gas_v4i_p)(uintptr_t)g,
        (lds_v4i_p)(unsigned int)(uintptr_t)l,
        0, 0);
#else
    *(uint4*)l = *(const uint4*)g;
#endif
}

// ---------------- Phase 1a: x (f32) -> Xb (bf16, row-major [M,K]) ----------
__global__ void convert_x_kernel(const float* __restrict__ x,
                                 unsigned short* __restrict__ xb, int n) {
    int i = blockIdx.x * blockDim.x + threadIdx.x;
    if (i < n) xb[i] = f32_to_bf16(x[i]);
}

// ---------------- Phase 1b: W [N,K] f32 -> Wt [K,N] bf16 (tiled transpose) -
__global__ void transpose_w_kernel(const float* __restrict__ W,
                                   unsigned short* __restrict__ Wt) {
    __shared__ float tile[32][33];
    int k0 = blockIdx.x * 32;   // K dimension
    int n0 = blockIdx.y * 32;   // N dimension
    int tx = threadIdx.x, ty = threadIdx.y;
    #pragma unroll
    for (int i = ty; i < 32; i += 8)
        tile[i][tx] = W[(size_t)(n0 + i) * K_ + (k0 + tx)];   // coalesced over K
    __syncthreads();
    #pragma unroll
    for (int i = ty; i < 32; i += 8)
        Wt[(size_t)(k0 + i) * N_ + (n0 + tx)] = f32_to_bf16(tile[tx][i]); // coalesced over N
}

// ---------------- Phase 2: gates = Xb @ Wt + b, bf16 WMMA ------------------
// Grid: (N_/128, M_/128); block: 128 threads = 4 wave32 in 2x2, wave tile 64x64.
// Double-buffered LDS; tile i+1 streams in via async-LDS while tile i runs
// through 16 v_wmma_f32_16x16x32_bf16.
__global__ void __launch_bounds__(128)
gates_gemm_kernel(const unsigned short* __restrict__ Xb,
                  const unsigned short* __restrict__ Wt,
                  const float* __restrict__ bias,
                  float* __restrict__ gates) {
    // Rows padded to dodge bank conflicts (strides in u16 units).
    __shared__ __align__(16) unsigned short aT[2][128 * 40];  // 128x32 bf16, 80B rows
    __shared__ __align__(16) unsigned short bT[2][32 * 136];  // 32x128 bf16, 272B rows

    const int tid = threadIdx.x;
    const int L   = tid & 31;          // lane in wave
    const int wv  = tid >> 5;          // wave id 0..3
    const int wM  = (wv >> 1) * 64;    // wave M offset in WG tile
    const int wN  = (wv & 1) * 64;     // wave N offset in WG tile
    const int m0  = blockIdx.y * 128;
    const int n0  = blockIdx.x * 128;

    // Issue one K-step tile pair (8 x B128 per thread => 8 async ops per wave).
    auto issue_tiles = [&](unsigned short* a, unsigned short* bb, int k0) {
        #pragma unroll
        for (int p = 0; p < 4; ++p) {          // A: 128 rows x 64B, 4 thr/row
            int row = (tid >> 2) + p * 32, ch = tid & 3;
            copy16_g2l(Xb + (size_t)(m0 + row) * K_ + k0 + ch * 8,
                       a + row * 40 + ch * 8);
        }
        #pragma unroll
        for (int p = 0; p < 4; ++p) {          // B: 32 rows x 256B, 16 thr/row
            int row = (tid >> 4) + p * 8, ch = tid & 15;
            copy16_g2l(Wt + (size_t)(k0 + row) * N_ + n0 + ch * 8,
                       bb + row * 136 + ch * 8);
        }
    };

    // Accumulators initialized with bias (column-broadcast across C rows).
    v8f acc[4][4];
    #pragma unroll
    for (int nf = 0; nf < 4; ++nf) {
        float bv = bias[n0 + wN + nf * 16 + (L & 15)];
        #pragma unroll
        for (int mf = 0; mf < 4; ++mf) {
            #pragma unroll
            for (int j = 0; j < 8; ++j) acc[mf][nf][j] = bv;
        }
    }

    issue_tiles(aT[0], bT[0], 0);              // prologue: tile 0 in flight

    for (int i = 0; i < K_ / 32; ++i) {
        const int cur = i & 1;
        const int k0  = i * 32;
        if (k0 + 32 < K_) {
            issue_tiles(aT[cur ^ 1], bT[cur ^ 1], k0 + 32);   // next tile in flight
#if USE_ASYNC_LDS
            __builtin_amdgcn_s_wait_asynccnt(8);   // current tile's 8 ops landed
#endif
        } else {
#if USE_ASYNC_LDS
            __builtin_amdgcn_s_wait_asynccnt(0);
#endif
        }
        __syncthreads();                       // all 4 waves see current tile

        const unsigned short* aBuf = aT[cur];
        const unsigned short* bBuf = bT[cur];

        // Build fragments per ISA 7.12.2 layouts.
        v16bf Af[4], Bf[4];
        #pragma unroll
        for (int mf = 0; mf < 4; ++mf) {
            // A 16x32: lane row = M; lanes<16 hold K 0-7 & 16-23, lanes>=16 K 8-15 & 24-31
            int lrow = wM + mf * 16 + (L & 15);
            int kb   = (L >> 4) * 8;                 // K element offset 0 or 8
            union { uint4 q[2]; v16bf v; } ua;
            ua.q[0] = *(const uint4*)(aBuf + lrow * 40 + kb);       // K = kb .. kb+7
            ua.q[1] = *(const uint4*)(aBuf + lrow * 40 + kb + 16);  // K = kb+16 .. kb+23
            Af[mf] = ua.v;
        }
        #pragma unroll
        for (int nf = 0; nf < 4; ++nf) {
            // B 32x16: lane col = N; lanes<16 hold K 0-7 & 16-23, lanes>=16 K 8-15 & 24-31
            int bcol = wN + nf * 16 + (L & 15);
            int kbb  = (L >> 4) * 8;
            union { unsigned short u[16]; v16bf v; } ub;
            #pragma unroll
            for (int s = 0; s < 16; ++s)
                ub.u[s] = bBuf[(kbb + (s < 8 ? s : s + 8)) * 136 + bcol];
            Bf[nf] = ub.v;
        }
        #pragma unroll
        for (int mf = 0; mf < 4; ++mf) {
            #pragma unroll
            for (int nf = 0; nf < 4; ++nf) {
                acc[mf][nf] = __builtin_amdgcn_wmma_f32_16x16x32_bf16(
                    false, Af[mf], false, Bf[nf], (short)0, acc[mf][nf],
                    false, false);
            }
        }
        __syncthreads();   // all reads of buf[cur] done before it is refilled
    }

    // Epilogue: C 16x16 f32 layout — lanes<16: VGPR j -> M=j; lanes>=16: M=j+8.
    #pragma unroll
    for (int mf = 0; mf < 4; ++mf) {
        size_t rbase = (size_t)(m0 + wM + mf * 16 + ((L >> 4) << 3));
        #pragma unroll
        for (int nf = 0; nf < 4; ++nf) {
            int col = n0 + wN + nf * 16 + (L & 15);
            #pragma unroll
            for (int j = 0; j < 8; ++j)
                gates[(rbase + j) * (size_t)N_ + col] = acc[mf][nf][j];
        }
    }
}

// ---------------- Phase 3: fused activations + sequential scan -------------
// One thread per (b,d); gate reads coalesced over d; prefetch runs ahead.
__global__ void __launch_bounds__(256)
scan_kernel(const float* __restrict__ gates, const float* __restrict__ h0,
            const float* __restrict__ s0, float* __restrict__ out) {
    int gid = blockIdx.x * 256 + threadIdx.x;   // 0 .. B*D-1 = 8191
    int bb  = gid >> 10;
    int d   = gid & (D_ - 1);
    float h = h0[gid];
    float s = s0[gid];
    size_t base  = (size_t)bb * T_ * N_ + d;
    size_t obase = (size_t)bb * T_ * D_ + d;
    float* sT = out + (size_t)B_ * T_ * D_;     // final state appended after out

    for (int t = 0; t < T_; ++t) {
        const float* g = gates + base;
        __builtin_prefetch(g + 2 * N_, 0, 0);   // global_prefetch_b8, 2 rows ahead
        float r  = g[0 * D_], z  = g[1 * D_], k  = g[2 * D_], v = g[3 * D_];
        float hp = g[4 * D_], dc = g[5 * D_], al = g[6 * D_];

        float rc2 = (r * r + z * z) * 0.5f + EPS_;
        float zs  = 1.0f / (1.0f + __expf(-z * rsqrtf(rc2)));
        float kv2 = (k * k + v * v) * 0.5f + EPS_;
        float kv  = (k * v) / (kv2 * (kv2 + EPS_));
        float hpn = hp * rsqrtf(hp * hp + EPS_);
        float dt  = 1.0f / (1.0f + __expf(-dc * rsqrtf(dc * dc + EPS_)));
        float at  = 1.0f / (1.0f + __expf(-al * rsqrtf(al * al + EPS_)));

        s = dt * s + at * kv;
        float c = tanhf(hpn + s);               // sigmoid(2x)*2-1 == tanh(x)
        h = (1.0f - zs) * c + zs * h;

        out[obase] = h;
        base  += N_;
        obase += D_;
    }
    sT[gid] = s;
}

// ---------------------------------------------------------------------------
extern "C" void kernel_launch(void* const* d_in, const int* in_sizes, int n_in,
                              void* d_out, int out_size, void* d_ws, size_t ws_size,
                              hipStream_t stream) {
    const float* x  = (const float*)d_in[0];
    const float* h0 = (const float*)d_in[1];
    const float* s0 = (const float*)d_in[2];
    const float* W  = (const float*)d_in[3];
    const float* b  = (const float*)d_in[4];
    float* out = (float*)d_out;

    unsigned char* ws = (unsigned char*)d_ws;
    unsigned short* Xb = (unsigned short*)ws;                                   // 32 MiB
    unsigned short* Wt = (unsigned short*)(ws + (size_t)M_ * K_ * 2);           // 14 MiB
    float* gates = (float*)(ws + (size_t)M_ * K_ * 2 + (size_t)K_ * N_ * 2);    // 448 MiB

    convert_x_kernel<<<(M_ * K_) / 256, 256, 0, stream>>>(x, Xb, M_ * K_);
    transpose_w_kernel<<<dim3(K_ / 32, N_ / 32), dim3(32, 8), 0, stream>>>(W, Wt);
    gates_gemm_kernel<<<dim3(N_ / 128, M_ / 128), 128, 0, stream>>>(Xb, Wt, b, gates);
    scan_kernel<<<(B_ * D_) / 256, 256, 0, stream>>>(gates, h0, s0, out);
}
